// InvariantPointAttention_32736240730456
// MI455X (gfx1250) — compile-verified, matrix operand order checked by reading, not run on hardware
//
#include <hip/hip_runtime.h>
#include <math.h>

typedef __bf16 bf16_t;
typedef __attribute__((ext_vector_type(16))) __bf16 v16bf;
typedef __attribute__((ext_vector_type(8)))  __bf16 v8bf;
typedef __attribute__((ext_vector_type(8)))  float  v8f;

constexpr int NN   = 4096;
constexpr int KN   = 24;
constexpr int CIN  = 256;   // IFZ
constexpr int AHh  = 12;
constexpr int AFd  = 48;
constexpr int PCOLS = 2304; // 576*3 + 144*2 + 288
constexpr int CATC  = 2496; // 576 + 1536 + 288 + 96

// column offsets inside P rows (concatenated projections)
constexpr int OQ1 = 0, OK1 = 576, OV1 = 1152, OQ2 = 1728, OK2 = 1872, OV2 = 2016;
// column offsets inside cat rows
constexpr int CO1 = 0, CO2 = 576, COI = 2112, CON = 2400;

// ---------------------------------------------------------------------------
// Direct-from-memory WMMA fragment loader (works for global or LDS pointers).
// Operand is stored row-major [16 rows][stride] bf16 with the 32-wide K run
// contiguous. CDNA5 16-bit A/B layout (ISA 7.12.2): lane = row(0..15) +
// 16*khalf; VGPRs 0-3 hold K = khalf*8..+7, VGPRs 4-7 hold K = 16+khalf*8..+7.
// Each half is 8 consecutive bf16 = one 16-byte (b128) load.
// ---------------------------------------------------------------------------
__device__ __forceinline__ v16bf load_frag_g(const bf16_t* base, int stride) {
  const int lane = threadIdx.x & 31;
  const int row  = lane & 15;
  const int kh   = lane >> 4;
  const bf16_t* p0 = base + (size_t)row * stride + kh * 8;  // K = kh*8 ..
  const v8bf lo = *(const v8bf*)p0;                          // 16-B aligned
  const v8bf hi = *(const v8bf*)(p0 + 16);                   // K = 16+kh*8 ..
  v16bf f;
#pragma unroll
  for (int j = 0; j < 8; ++j) { f[j] = lo[j]; f[8 + j] = hi[j]; }
  return f;
}

// ---------------------------------------------------------------------------
// Kernel 0: pack operands into WMMA-native bf16 layouts (runs every launch;
// deterministic). x1b[4096][256], WcatT[2304 n][256 k], WbT[256 n][2496 k].
// ---------------------------------------------------------------------------
constexpr int NPACK_X1 = NN * CIN;          // 1,048,576
constexpr int NPACK_WC = PCOLS * CIN;       //   589,824
constexpr int NPACK_WB = CIN * CATC;        //   638,976
constexpr int NPACK    = NPACK_X1 + NPACK_WC + NPACK_WB;

__device__ __forceinline__ float wsel(
    int gc, int kr,
    const float* Wq1, const float* Wk1, const float* Wv1,
    const float* Wq2, const float* Wk2, const float* Wv2)
{
  if (gc <  576) return Wq1[kr * 576 + gc];
  if (gc < 1152) return Wk1[kr * 576 + (gc - 576)];
  if (gc < 1728) return Wv1[kr * 576 + (gc - 1152)];
  if (gc < 1872) return Wq2[kr * 144 + (gc - 1728)];
  if (gc < 2016) return Wk2[kr * 144 + (gc - 1872)];
  return Wv2[kr * 288 + (gc - 2016)];
}

__global__ __launch_bounds__(256) void pack_kernel(
    const float* __restrict__ x1,
    const float* __restrict__ Wq1, const float* __restrict__ Wk1,
    const float* __restrict__ Wv1, const float* __restrict__ Wq2,
    const float* __restrict__ Wk2, const float* __restrict__ Wv2,
    const float* __restrict__ Wback,
    bf16_t* __restrict__ x1b, bf16_t* __restrict__ WcatT,
    bf16_t* __restrict__ WbT)
{
  const int idx = blockIdx.x * 256 + threadIdx.x;
  if (idx < NPACK_X1) {
    x1b[idx] = (bf16_t)x1[idx];
  } else if (idx < NPACK_X1 + NPACK_WC) {
    const int i = idx - NPACK_X1;
    const int n = i >> 8, k = i & 255;           // WcatT[n][k] = Wcat[k][n]
    WcatT[i] = (bf16_t)wsel(n, k, Wq1, Wk1, Wv1, Wq2, Wk2, Wv2);
  } else if (idx < NPACK) {
    const int i = idx - NPACK_X1 - NPACK_WC;
    const int n = i / CATC, k = i - n * CATC;    // WbT[n][k] = Wback[k][n]
    WbT[i] = (bf16_t)Wback[(size_t)k * CIN + n];
  }
}

// ---------------------------------------------------------------------------
// Kernel 1: P[4096 x 2304] = x1 @ Wcat. Block tile 32x64 = 8 waves x one
// 16x16 tile. No LDS, no barriers: fragments load straight from L2-resident
// bf16 operands (layout == fragment layout). 8 WMMA per wave.
// ---------------------------------------------------------------------------
__global__ __launch_bounds__(256) void proj_kernel(
    const bf16_t* __restrict__ x1b, const bf16_t* __restrict__ WcatT,
    float* __restrict__ P)
{
  const int wave = threadIdx.x >> 5;
  const int rowT = blockIdx.x * 2 + (wave & 1);   // 16-row tile
  const int colT = blockIdx.y * 4 + (wave >> 1);  // 16-col tile
  const bf16_t* Abase = x1b   + (size_t)rowT * 16 * CIN;
  const bf16_t* Bbase = WcatT + (size_t)colT * 16 * CIN;

  v8f acc = {};
#pragma unroll
  for (int kk = 0; kk < CIN; kk += 32) {
    const v16bf a = load_frag_g(Abase + kk, CIN);
    const v16bf b = load_frag_g(Bbase + kk, CIN);
    acc = __builtin_amdgcn_wmma_f32_16x16x32_bf16(
        false, a, false, b, (short)0, acc, false, false);
  }

  // C/D layout: col = lane&15, VGPR r -> row r + 8*(lane>>4)
  const int lane  = threadIdx.x & 31;
  const int col   = colT * 16 + (lane & 15);
  const int rbase = (lane >> 4) * 8;
#pragma unroll
  for (int r = 0; r < 8; ++r)
    P[(size_t)(rowT * 16 + rbase + r) * PCOLS + col] = acc[r];
}

// ---------------------------------------------------------------------------
// Kernel 2: v2t[n, a, v, :] = R_n @ v2[n,a,v,:] + t_n   (transform-then-gather)
// ---------------------------------------------------------------------------
__global__ __launch_bounds__(96) void v2t_kernel(
    const float* __restrict__ P, const float* __restrict__ aff,
    float* __restrict__ v2t)
{
  const int n = blockIdx.x;
  const int t = threadIdx.x;            // (a,v) pair, 0..95
  const float* A = aff + n * 12;        // rows: [R00 R01 R02 t0] ...
  float R[9], tr[3];
#pragma unroll
  for (int i = 0; i < 3; ++i) {
#pragma unroll
    for (int j = 0; j < 3; ++j) R[i * 3 + j] = A[i * 4 + j];
    tr[i] = A[i * 4 + 3];
  }
  const float* v = P + (size_t)n * PCOLS + OV2 + t * 3;
  const float x = v[0], y = v[1], z = v[2];
#pragma unroll
  for (int i = 0; i < 3; ++i)
    v2t[(size_t)n * 288 + t * 3 + i] =
        R[i * 3 + 0] * x + R[i * 3 + 1] * y + R[i * 3 + 2] * z + tr[i];
}

// ---------------------------------------------------------------------------
// Kernel 3: attention per node. One block (8 waves) per node. Gathers hit L2
// (P is 38 MB, L2 is 192 MB). x2 row staged via ASYNC global->LDS copy
// (ASYNCcnt path, CDNA5 ISA). Writes cat row in bf16.
// ---------------------------------------------------------------------------
__global__ __launch_bounds__(256) void attn_kernel(
    const float* __restrict__ P,    const float* __restrict__ x2,
    const float* __restrict__ aff,  const float* __restrict__ pos,
    const int*   __restrict__ edge, const float* __restrict__ Wbia,
    const float* __restrict__ gam,  const float* __restrict__ v2t,
    bf16_t* __restrict__ cat)
{
  __shared__ float q1r[AHh * AFd];     // roped q1
  __shared__ float q2l[144];           // raw q2 (local coords)
  __shared__ float x2s[KN * 128];
  __shared__ float sc [KN * AHh];
  __shared__ float wgt[KN * AHh];
  __shared__ float Rn[9], tn[3];
  __shared__ int   je[KN];

  const int n = blockIdx.x;
  const int t = threadIdx.x;

  // ---- async-stage x2 row (24x128 f32 = 12 KB) into LDS ------------------
  {
    const float* src = x2 + (size_t)n * KN * 128;
    const unsigned ldsbase = (unsigned)(uintptr_t)(void*)x2s;
    for (int idx = t; idx < (KN * 128) / 4; idx += 256) {   // 768 x b128
      const unsigned           lds = ldsbase + idx * 16u;
      const unsigned long long ga  = (unsigned long long)(const void*)(src + idx * 4);
      asm volatile("global_load_async_to_lds_b128 %0, %1, off"
                   :: "v"(lds), "v"(ga) : "memory");
    }
    asm volatile("s_wait_asynccnt 0x0" ::: "memory");
  }

  // ---- stage the rest -----------------------------------------------------
  if (t < 9)               Rn[t]     = aff[n * 12 + (t / 3) * 4 + (t % 3)];
  else if (t < 12)         tn[t - 9] = aff[n * 12 + (t - 9) * 4 + 3];
  if (t < KN)              je[t]     = edge[n * KN + t];
  if (t < 144)             q2l[t]    = P[(size_t)n * PCOLS + OQ2 + t];
  for (int idx = t; idx < AHh * 24; idx += 256) {  // rope q1: (a, freq f)
    const int a = idx / 24, f = idx % 24;
    float c, s; __sincosf(pos[n * 24 + f], &s, &c);
    const float* qp = P + (size_t)n * PCOLS + OQ1 + a * AFd + 2 * f;
    const float e = qp[0], o = qp[1];
    q1r[a * AFd + 2 * f + 0] = e * c - o * s;
    q1r[a * AFd + 2 * f + 1] = e * s + o * c;
  }
  __syncthreads();

  // ---- scores -------------------------------------------------------------
  const float wl  = 0.57735026919f;    // sqrt(1/3)
  const float wc  = 0.23570226040f;    // sqrt(2/(9*QP)) = sqrt(2/36)
  const float isq = 0.14433756730f;    // 1/sqrt(48)
  for (int idx = t; idx < KN * AHh; idx += 256) {
    const int k = idx / AHh, a = idx % AHh;
    const int j = je[k];
    // s1: rope gathered k1 on the fly, dot with roped q1
    float s1 = 0.f;
    const float* k1p  = P + (size_t)j * PCOLS + OK1 + a * AFd;
    const float* posj = pos + j * 24;
#pragma unroll
    for (int f = 0; f < 24; ++f) {
      float c, s; __sincosf(posj[f], &s, &c);
      const float e = k1p[2 * f], o = k1p[2 * f + 1];
      s1 += q1r[a * AFd + 2 * f]     * (e * c - o * s)
          + q1r[a * AFd + 2 * f + 1] * (e * s + o * c);
    }
    s1 *= isq;
    // bias = x2[n,k,:] . Wbia[:,a]
    float bi = 0.f;
    for (int i = 0; i < 128; ++i) bi += x2s[k * 128 + i] * Wbia[i * AHh + a];
    // s2: -|R_n (q2 - k2)|^2  (translation cancels)
    float s2 = 0.f;
    const float* k2p = P + (size_t)j * PCOLS + OK2 + a * 12;
#pragma unroll
    for (int p = 0; p < 4; ++p) {
      const float dx = q2l[a * 12 + p * 3 + 0] - k2p[p * 3 + 0];
      const float dy = q2l[a * 12 + p * 3 + 1] - k2p[p * 3 + 1];
      const float dz = q2l[a * 12 + p * 3 + 2] - k2p[p * 3 + 2];
#pragma unroll
      for (int i = 0; i < 3; ++i) {
        const float d = Rn[i*3+0] * dx + Rn[i*3+1] * dy + Rn[i*3+2] * dz;
        s2 -= d * d;
      }
    }
    sc[k * AHh + a] = wl * (s1 + bi + 0.1f * wc * gam[a] * s2);
  }
  __syncthreads();

  // ---- softmax over K=24 (one thread per head) ---------------------------
  if (t < AHh) {
    float m = -INFINITY;
    for (int k = 0; k < KN; ++k) m = fmaxf(m, sc[k * AHh + t]);
    float s = 0.f;
    for (int k = 0; k < KN; ++k) {
      const float e = __expf(sc[k * AHh + t] - m);
      wgt[k * AHh + t] = e; s += e;
    }
    const float inv = 1.f / s;
    for (int k = 0; k < KN; ++k) wgt[k * AHh + t] *= inv;
  }
  __syncthreads();

  bf16_t* crow = cat + (size_t)n * CATC;
  // ---- out1 = sum_k w * v1_gathered --------------------------------------
  for (int idx = t; idx < AHh * AFd; idx += 256) {
    const int a = idx / AFd, i = idx % AFd;
    float acc = 0.f;
    for (int k = 0; k < KN; ++k)
      acc += wgt[k * AHh + a] * P[(size_t)je[k] * PCOLS + OV1 + a * AFd + i];
    crow[CO1 + idx] = (bf16_t)acc;
  }
  // ---- out2 = sum_k w * x2 -----------------------------------------------
  for (int idx = t; idx < AHh * 128; idx += 256) {
    const int a = idx >> 7, i = idx & 127;
    float acc = 0.f;
    for (int k = 0; k < KN; ++k) acc += wgt[k * AHh + a] * x2s[k * 128 + i];
    crow[CO2 + idx] = (bf16_t)acc;
  }
  // ---- oipa = R^T (sum_k w * v2t_gathered - t), onorm --------------------
  if (t < 96) {
    const int a = t / 8;
    float ax = 0.f, ay = 0.f, az = 0.f;
    for (int k = 0; k < KN; ++k) {
      const float* vp = v2t + (size_t)je[k] * 288 + t * 3;
      const float w = wgt[k * AHh + a];
      ax += w * vp[0]; ay += w * vp[1]; az += w * vp[2];
    }
    ax -= tn[0]; ay -= tn[1]; az -= tn[2];
    const float o0 = Rn[0] * ax + Rn[3] * ay + Rn[6] * az;
    const float o1 = Rn[1] * ax + Rn[4] * ay + Rn[7] * az;
    const float o2 = Rn[2] * ax + Rn[5] * ay + Rn[8] * az;
    crow[COI + t * 3 + 0] = (bf16_t)o0;
    crow[COI + t * 3 + 1] = (bf16_t)o1;
    crow[COI + t * 3 + 2] = (bf16_t)o2;
    crow[CON + t] = (bf16_t)sqrtf(o0 * o0 + o1 * o1 + o2 * o2);
  }
}

// ---------------------------------------------------------------------------
// Kernel 4: out = LayerNorm( sqrt(2)*x1 + cat @ Wback + bback )
// Block: 16 rows x full 256 cols; wave w owns col tiles w*32, w*32+16.
// Fragments load direct from bf16 global (cat rows / WbT rows); no barriers
// in the 78-step K loop. LDS only for the LayerNorm reduction.
// ---------------------------------------------------------------------------
__global__ __launch_bounds__(256) void back_ln_kernel(
    const bf16_t* __restrict__ cat,  const bf16_t* __restrict__ WbT,
    const float*  __restrict__ bback,const float* __restrict__ x1,
    const float*  __restrict__ ln_g, const float* __restrict__ ln_b,
    float* __restrict__ out)
{
  __shared__ float hs[16][256];
  __shared__ float mu_s[16], rs_s[16];

  const int t    = threadIdx.x;
  const int wave = t >> 5;
  const int rowT = blockIdx.x;

  const bf16_t* Abase  = cat + (size_t)rowT * 16 * CATC;
  const bf16_t* B0base = WbT + (size_t)(wave * 32 +  0) * CATC;
  const bf16_t* B1base = WbT + (size_t)(wave * 32 + 16) * CATC;

  v8f acc0 = {}, acc1 = {};
  for (int kk = 0; kk < CATC; kk += 32) {   // 78 steps
    const v16bf a  = load_frag_g(Abase  + kk, CATC);
    const v16bf b0 = load_frag_g(B0base + kk, CATC);
    const v16bf b1 = load_frag_g(B1base + kk, CATC);
    acc0 = __builtin_amdgcn_wmma_f32_16x16x32_bf16(
        false, a, false, b0, (short)0, acc0, false, false);
    acc1 = __builtin_amdgcn_wmma_f32_16x16x32_bf16(
        false, a, false, b1, (short)0, acc1, false, false);
  }

  const int lane = t & 31;
  const int cl   = lane & 15;
  const int rb   = (lane >> 4) * 8;
#pragma unroll
  for (int r = 0; r < 8; ++r) {
    hs[rb + r][wave * 32 + cl]      = acc0[r];
    hs[rb + r][wave * 32 + 16 + cl] = acc1[r];
  }
  __syncthreads();

  // bias + residual
  for (int idx = t; idx < 16 * 256; idx += 256) {
    const int r = idx >> 8, c = idx & 255;
    hs[r][c] = 1.41421356237f * x1[(size_t)(rowT * 16 + r) * CIN + c]
             + hs[r][c] + bback[c];
  }
  __syncthreads();

  if (t < 16) {
    float s = 0.f, s2 = 0.f;
    for (int c = 0; c < 256; ++c) { const float h = hs[t][c]; s += h; s2 += h * h; }
    const float mu  = s * (1.f / 256.f);
    const float var = s2 * (1.f / 256.f) - mu * mu;
    mu_s[t] = mu;
    rs_s[t] = rsqrtf(var + 1e-5f);
  }
  __syncthreads();

  for (int idx = t; idx < 16 * 256; idx += 256) {
    const int r = idx >> 8, c = idx & 255;
    out[(size_t)(rowT * 16 + r) * CIN + c] =
        (hs[r][c] - mu_s[r]) * rs_s[r] * ln_g[c] + ln_b[c];
  }
}

// ---------------------------------------------------------------------------
extern "C" void kernel_launch(void* const* d_in, const int* in_sizes, int n_in,
                              void* d_out, int out_size, void* d_ws, size_t ws_size,
                              hipStream_t stream)
{
  const float* x1    = (const float*)d_in[0];
  const float* x2    = (const float*)d_in[1];
  const float* aff   = (const float*)d_in[2];
  const float* pos   = (const float*)d_in[3];
  const int*   edge  = (const int*)  d_in[4];
  const float* Wq1   = (const float*)d_in[5];
  const float* Wk1   = (const float*)d_in[6];
  const float* Wv1   = (const float*)d_in[7];
  const float* Wq2   = (const float*)d_in[8];
  const float* Wk2   = (const float*)d_in[9];
  const float* Wv2   = (const float*)d_in[10];
  const float* Wbia  = (const float*)d_in[11];
  const float* gam   = (const float*)d_in[12];
  const float* Wback = (const float*)d_in[13];
  const float* bback = (const float*)d_in[14];
  const float* ln_g  = (const float*)d_in[15];
  const float* ln_b  = (const float*)d_in[16];
  float* out = (float*)d_out;

  char* ws = (char*)d_ws;
  size_t off = 0;
  float*  P     = (float*)(ws + off);  off += (size_t)NN * PCOLS * sizeof(float);   // 37.75 MB
  float*  v2t   = (float*)(ws + off);  off += (size_t)NN * 288 * sizeof(float);     //  4.72 MB
  bf16_t* cat   = (bf16_t*)(ws + off); off += (size_t)NN * CATC * sizeof(bf16_t);   // 20.45 MB
  bf16_t* x1b   = (bf16_t*)(ws + off); off += (size_t)NPACK_X1 * sizeof(bf16_t);    //  2.00 MB
  bf16_t* WcatT = (bf16_t*)(ws + off); off += (size_t)NPACK_WC * sizeof(bf16_t);    //  1.13 MB
  bf16_t* WbT   = (bf16_t*)(ws + off); off += (size_t)NPACK_WB * sizeof(bf16_t);    //  1.22 MB

  pack_kernel<<<(NPACK + 255) / 256, 256, 0, stream>>>(
      x1, Wq1, Wk1, Wv1, Wq2, Wk2, Wv2, Wback, x1b, WcatT, WbT);
  proj_kernel<<<dim3(NN / 32, PCOLS / 64), 256, 0, stream>>>(x1b, WcatT, P);
  v2t_kernel<<<NN, 96, 0, stream>>>(P, aff, v2t);
  attn_kernel<<<NN, 256, 0, stream>>>(P, x2, aff, pos, edge, Wbia, gam, v2t, cat);
  back_ln_kernel<<<NN / 16, 256, 0, stream>>>(cat, WbT, bback, x1, ln_g, ln_b, out);
}
